// RyeModel_56805237457519
// MI455X (gfx1250) — compile-verified, matrix operand order checked by reading, not run on hardware
//
#include <hip/hip_runtime.h>
#include <hip/hip_bf16.h>
#include <math.h>

// ---------------------------------------------------------------------------
// RyeModel on gfx1250 (MI455X):
//  kernel 1: Gumbel-max random walks (wave-per-walk, lane-parallel argmax)
//  kernel 2: recurrent cell, 8 steps, dual bf16 WMMA GEMMs per step.
//            Weights staged in LDS in WMMA-fragment-major layout so every
//            B operand is two ds_load_b128 straight into the operand regs.
// ---------------------------------------------------------------------------

typedef __attribute__((ext_vector_type(16))) __bf16        v16bf;
typedef __attribute__((ext_vector_type(8)))  float         v8f;
typedef __attribute__((ext_vector_type(8)))  unsigned int  v8u;

#define LENGTH   8
#define REPEAT   4
#define HIDDEN   64
#define CHANNELS 16
#define FEAT     64
#define KDIM     144          // FEAT + HIDDEN + CHANNELS
#define KPAD     160          // padded to 5 chunks of 32 for wmma 16x16x32
#define THREADS  128          // 4 waves per workgroup, 16 rows per wave

// float -> bf16, round-to-nearest-even
__device__ __forceinline__ unsigned short f2bf(float f) {
    unsigned int u = __builtin_bit_cast(unsigned int, f);
    u += 0x7FFFu + ((u >> 16) & 1u);
    return (unsigned short)(u >> 16);
}

__device__ __forceinline__ unsigned int mix32(unsigned int x) {
    x ^= x >> 16; x *= 0x7FEB352Du;
    x ^= x >> 15; x *= 0x846CA68Bu;
    x ^= x >> 16;
    return x;
}

// fast activations: v_exp + v_rcp, no div_scale chains
__device__ __forceinline__ float fast_tanh(float x) {
    return 1.0f - 2.0f * __builtin_amdgcn_rcpf(1.0f + __expf(2.0f * x));
}
__device__ __forceinline__ float fast_sigmoid(float x) {
    return __builtin_amdgcn_rcpf(1.0f + __expf(-x));
}

// ---------------------------------------------------------------------------
// Kernel 1: random walks.  walks[(r*L + l)*N + n], flipped so slot L-1 = n.
// ---------------------------------------------------------------------------
__global__ __launch_bounds__(256) void rye_walks_kernel(
    const float* __restrict__ prob, int* __restrict__ walks, int N)
{
    const int wid  = (blockIdx.x * blockDim.x + threadIdx.x) >> 5;
    const int lane = threadIdx.x & 31;
    if (wid >= REPEAT * N) return;
    const int r = wid / N;
    const int n = wid - r * N;

    int cur = n;
    if (lane == 0) walks[(r * LENGTH + (LENGTH - 1)) * N + n] = cur;

    for (int s = 1; s < LENGTH; ++s) {
        const float* __restrict__ row = prob + (size_t)cur * N;
        float best = -3.0e38f;
        int   bidx = 0;
        for (int c = lane; c < N; c += 32) {
            if (c + 256 < N) __builtin_prefetch(row + c + 256, 0, 1);  // global_prefetch_b8
            unsigned int h = mix32((unsigned)n * 0x9E3779B9u ^
                                   (unsigned)r * 0x85EBCA6Bu ^
                                   (unsigned)s * 0xC2B2AE35u ^
                                   (unsigned)c * 0x27D4EB2Fu);
            float u = ((float)(h & 0x00FFFFFFu) + 0.5f) * (1.0f / 16777216.0f);
            float g = -__logf(-__logf(u));                 // Gumbel(0,1)
            float v = __logf(row[c] + 1e-9f) + g;
            if (v > best) { best = v; bidx = c; }
        }
        for (int off = 16; off > 0; off >>= 1) {           // butterfly argmax
            float ob = __shfl_xor(best, off, 32);
            int   oi = __shfl_xor(bidx, off, 32);
            if (ob > best || (ob == best && oi < bidx)) { best = ob; bidx = oi; }
        }
        cur = bidx;
        if (lane == 0) walks[(r * LENGTH + (LENGTH - 1 - s)) * N + n] = cur;
    }
}

// ---------------------------------------------------------------------------
// Kernel 2: 8-step recurrence.  4 waves x 16 rows = 64 rows per workgroup.
// ---------------------------------------------------------------------------
__global__ __launch_bounds__(THREADS) void rye_cell_kernel(
    const float* __restrict__ inv_input,   // [N,64]
    const float* __restrict__ eq_input,    // [N,3]
    const float* __restrict__ W1,          // [144,64]
    const float* __restrict__ b1,          // [64]
    const float* __restrict__ W2,          // [64,32]
    const float* __restrict__ b2,          // [32]
    const int*   __restrict__ walks,       // [R,L,N]
    float* __restrict__ inv_out,           // [R,L,N,64]
    float* __restrict__ eq_out,            // [R,L,N,3,16]
    int N)
{
    // Fragment-major weights: frag f = kc*NT + nt; dword f*256 + lane*8 + i
    // holds pack(bf16(W[k0,n]), bf16(W[k0+1,n])) with
    //   k0 = kc*32 + (i>>2)*16 + (lane>>4)*8 + (i&3)*2,  n = nt*16 + (lane&15)
    __shared__ unsigned int   W1f[20 * 256];                 // 20 KB (5 kc x 4 nt)
    __shared__ unsigned int   W2f[4 * 256];                  //  4 KB (2 kc x 2 nt)
    __shared__ unsigned short feats[4][16 * KPAD];           // 20 KB per-wave tiles
    __shared__ float          eqd[4][16][3];

    const int t    = threadIdx.x;
    const int w    = t >> 5;
    const int lane = t & 31;
    const int half = lane >> 4;
    const int lc   = lane & 15;
    const int blocksPerR = N / 64;
    const int r       = blockIdx.x / blocksPerR;
    const int rowbase = (blockIdx.x - r * blocksPerR) * 64 + w * 16;

    // ---- stage W1 fragments (zero-pad K 144..159) ----
    for (int idx = t; idx < 20 * 256; idx += THREADS) {
        int f = idx >> 8, rem = idx & 255;
        int ls = rem >> 3, i = rem & 7;
        int kc = f >> 2, nt = f & 3;
        int k0 = kc * 32 + ((i >> 2) << 4) + ((ls >> 4) << 3) + ((i & 3) << 1);
        int n  = (nt << 4) + (ls & 15);
        unsigned int lo = (k0     < KDIM) ? (unsigned)f2bf(W1[k0 * HIDDEN + n])       : 0u;
        unsigned int hi = (k0 + 1 < KDIM) ? (unsigned)f2bf(W1[(k0 + 1) * HIDDEN + n]) : 0u;
        W1f[idx] = lo | (hi << 16);
    }
    // ---- stage W2 fragments ----
    for (int idx = t; idx < 4 * 256; idx += THREADS) {
        int f = idx >> 8, rem = idx & 255;
        int ls = rem >> 3, i = rem & 7;
        int kc = f >> 1, nt = f & 1;
        int k0 = kc * 32 + ((i >> 2) << 4) + ((ls >> 4) << 3) + ((i & 3) << 1);
        int n  = (nt << 4) + (ls & 15);
        W2f[idx] = (unsigned)f2bf(W2[k0 * 32 + n]) |
                   ((unsigned)f2bf(W2[(k0 + 1) * 32 + n]) << 16);
    }
    // init hidden/sqn/pad region of feats (cols 64..159) to zero
    for (int idx = lane; idx < 16 * 96; idx += 32) {
        int m = idx / 96, k = 64 + idx % 96;
        feats[w][m * KPAD + k] = 0;
    }
    __syncthreads();

    float b1v[4], b2v[2];
    #pragma unroll
    for (int nt = 0; nt < 4; ++nt) b1v[nt] = b1[nt * 16 + lc];
    b2v[0] = b2[lc];
    b2v[1] = b2[16 + lc];

    float eqh[3][8];
    #pragma unroll
    for (int d = 0; d < 3; ++d)
        #pragma unroll
        for (int j = 0; j < 8; ++j) eqh[d][j] = 0.0f;

    v8f vzero;
    #pragma unroll
    for (int i = 0; i < 8; ++i) vzero[i] = 0.0f;

    for (int l = 0; l < LENGTH; ++l) {
        // ---- per-step eq displacement (lanes 0..15 own one row each) ----
        if (lane < 16) {
            int gn = rowbase + lane;
            if (l == 0) {
                eqd[w][lane][0] = 0.f; eqd[w][lane][1] = 0.f; eqd[w][lane][2] = 0.f;
            } else {
                int ncur  = walks[(r * LENGTH + l)     * N + gn];
                int nprev = walks[(r * LENGTH + l - 1) * N + gn];
                #pragma unroll
                for (int d = 0; d < 3; ++d)
                    eqd[w][lane][d] = eq_input[ncur * 3 + d] - eq_input[nprev * 3 + d];
            }
        }
        // ---- gather inv_input rows along the walk -> feats[:,0:64] bf16 ----
        for (int idx = lane; idx < 16 * 32; idx += 32) {
            int m = idx >> 5, pr = idx & 31;
            int node = walks[(r * LENGTH + l) * N + rowbase + m];
            float2 v = *(const float2*)&inv_input[(size_t)node * FEAT + pr * 2];
            *(unsigned int*)&feats[w][m * KPAD + pr * 2] =
                (unsigned)f2bf(v.x) | ((unsigned)f2bf(v.y) << 16);
        }
        __syncthreads();

        // ---- GEMM1: feats @ W1, 5 k-chunks x 4 n-tiles ----
        v8f acc[4];
        #pragma unroll
        for (int nt = 0; nt < 4; ++nt) acc[nt] = vzero;
        #pragma unroll
        for (int kc = 0; kc < 5; ++kc) {
            // A fragment: two aligned b128 LDS loads from the row-major tile
            const unsigned short* ap = &feats[w][lc * KPAD + kc * 32 + half * 8];
            uint4 a0 = *(const uint4*)ap;
            uint4 a1 = *(const uint4*)(ap + 16);
            v8u au; au[0]=a0.x; au[1]=a0.y; au[2]=a0.z; au[3]=a0.w;
                    au[4]=a1.x; au[5]=a1.y; au[6]=a1.z; au[7]=a1.w;
            v16bf a = __builtin_bit_cast(v16bf, au);
            #pragma unroll
            for (int nt = 0; nt < 4; ++nt) {
                // B fragment: contiguous 32B per lane (two b128 loads)
                v8u bu = *(const v8u*)&W1f[((kc * 4 + nt) << 8) + (lane << 3)];
                v16bf b = __builtin_bit_cast(v16bf, bu);
                acc[nt] = __builtin_amdgcn_wmma_f32_16x16x32_bf16(
                    false, a, false, b, (short)0, acc[nt], false, false);
            }
        }

        // ---- tanh, write inv_traj, feed inv_h back as bf16 A-operand ----
        const size_t obase = ((size_t)(r * LENGTH + l) * N + rowbase) * HIDDEN;
        #pragma unroll
        for (int nt = 0; nt < 4; ++nt) {
            #pragma unroll
            for (int j = 0; j < 8; ++j) {
                int m = half * 8 + j;
                float h = fast_tanh(acc[nt][j] + b1v[nt]);
                inv_out[obase + (size_t)m * HIDDEN + nt * 16 + lc] = h;
                feats[w][m * KPAD + FEAT + nt * 16 + lc] = f2bf(h);
            }
        }
        __syncthreads();

        // ---- GEMM2: inv_h @ W2, 2 k-chunks x 2 n-tiles ----
        v8f gsc[2];
        gsc[0] = vzero; gsc[1] = vzero;
        #pragma unroll
        for (int kc = 0; kc < 2; ++kc) {
            const unsigned short* ap = &feats[w][lc * KPAD + FEAT + kc * 32 + half * 8];
            uint4 a0 = *(const uint4*)ap;
            uint4 a1 = *(const uint4*)(ap + 16);
            v8u au; au[0]=a0.x; au[1]=a0.y; au[2]=a0.z; au[3]=a0.w;
                    au[4]=a1.x; au[5]=a1.y; au[6]=a1.z; au[7]=a1.w;
            v16bf a = __builtin_bit_cast(v16bf, au);
            #pragma unroll
            for (int nt = 0; nt < 2; ++nt) {
                v8u bu = *(const v8u*)&W2f[((kc * 2 + nt) << 8) + (lane << 3)];
                v16bf b = __builtin_bit_cast(v16bf, bu);
                gsc[nt] = __builtin_amdgcn_wmma_f32_16x16x32_bf16(
                    false, a, false, b, (short)0, gsc[nt], false, false);
            }
        }

        // ---- gate/scale, eq_h update, sqn feedback, write eq_traj ----
        const size_t ebase = ((size_t)(r * LENGTH + l) * N + rowbase) * (3 * CHANNELS);
        #pragma unroll
        for (int j = 0; j < 8; ++j) {
            int m = half * 8 + j;
            float gate  = fast_sigmoid(gsc[0][j] + b2v[0]);
            float scale = gsc[1][j] + b2v[1];
            float sq = 0.0f;
            #pragma unroll
            for (int d = 0; d < 3; ++d) {
                float e = eqh[d][j] * gate + eqd[w][m][d] * scale;
                eqh[d][j] = e;
                sq += e * e;
                eq_out[ebase + (size_t)m * (3 * CHANNELS) + d * CHANNELS + lc] = e;
            }
            feats[w][m * KPAD + FEAT + HIDDEN + lc] = f2bf(sq);  // sqn for next step
        }
        // next iteration's __syncthreads (after gather) orders the sqn writes
    }
}

// ---------------------------------------------------------------------------
extern "C" void kernel_launch(void* const* d_in, const int* in_sizes, int n_in,
                              void* d_out, int out_size, void* d_ws, size_t ws_size,
                              hipStream_t stream)
{
    const float* prob   = (const float*)d_in[0];   // [N,N]
    const float* inv_in = (const float*)d_in[1];   // [N,64]
    const float* eq_in  = (const float*)d_in[2];   // [N,3]
    const float* W1     = (const float*)d_in[3];   // [144,64]
    const float* b1     = (const float*)d_in[4];   // [64]
    const float* W2     = (const float*)d_in[5];   // [64,32]
    const float* b2     = (const float*)d_in[6];   // [32]

    const int N = in_sizes[2] / 3;                 // 4096

    int* walks = (int*)d_ws;                       // R*L*N ints = 512 KB scratch

    float* inv_out = (float*)d_out;                                        // [R,L,N,64]
    float* eq_out  = (float*)d_out + (size_t)REPEAT * LENGTH * N * HIDDEN; // [R,L,N,3,16]

    {   // kernel 1: one wave per (r,n) walk
        int waves   = REPEAT * N;
        int threads = waves * 32;
        dim3 block(256);
        dim3 grid((threads + 255) / 256);
        rye_walks_kernel<<<grid, block, 0, stream>>>(prob, walks, N);
    }
    {   // kernel 2: 64 rows per workgroup
        dim3 block(THREADS);
        dim3 grid(REPEAT * (N / 64));
        rye_cell_kernel<<<grid, block, 0, stream>>>(
            inv_in, eq_in, W1, b1, W2, b2, walks, inv_out, eq_out, N);
    }
}
